// MyCrossAttentionLayer_89361089561219
// MI455X (gfx1250) — matrix operand order repaired and checked by reading, not compiled, not run on hardware
//
#include <hip/hip_runtime.h>
#include <hip/hip_bf16.h>
#include <cstdint>

#define D_MODEL 256
#define N_HEAD  4
#define LEN_Q   1024
#define LEN_K   4096
#define BATCH   8

typedef __attribute__((ext_vector_type(16))) __bf16         bf16x16;
typedef __attribute__((ext_vector_type(8)))  float          v8f;
typedef __attribute__((ext_vector_type(8)))  unsigned short v8us;
typedef __attribute__((ext_vector_type(4)))  unsigned int   v4u;
typedef __attribute__((ext_vector_type(8)))  int            v8i;
typedef __attribute__((ext_vector_type(4)))  int            v4i;

#if __has_builtin(__builtin_amdgcn_tensor_load_to_lds)
#define HAVE_TDM 1
#else
#define HAVE_TDM 0
#endif

union Frag {
    bf16x16      v;
    unsigned int u[8];
};

__device__ __forceinline__ unsigned short f2bf(float f) {
    unsigned int u = __float_as_uint(f);
    u += 0x7FFFu + ((u >> 16) & 1u);   // round-to-nearest-even
    return (unsigned short)(u >> 16);
}

__device__ __forceinline__ v8f wmma_bf16(const Frag& a, const Frag& b, v8f c) {
    return __builtin_amdgcn_wmma_f32_16x16x32_bf16(
        /*neg_a=*/false, a.v, /*neg_b=*/false, b.v,
        /*c_mod=*/(short)0, c, /*reuse_a=*/false, /*reuse_b=*/false);
}

// A fragment: 16x32 bf16 tile from row-major [.., ld] at (row0, k0).
__device__ __forceinline__ void load_a(Frag& f, const unsigned short* base, int ld,
                                       int row0, int k0) {
    int lane = threadIdx.x & 31;
    int m = lane & 15, half = lane >> 4;
    const unsigned short* p = base + (size_t)(row0 + m) * ld + k0;
#pragma unroll
    for (int j = 0; j < 8; ++j) {
        int kk = (j < 4) ? (8 * half + 2 * j) : (16 + 8 * half + 2 * (j - 4));
        f.u[j] = *(const unsigned int*)(p + kk);
    }
}

// B fragment (32x16) where B[k][n] = base[(nrow0+n)*ld + k0 + k].
__device__ __forceinline__ void load_bt(Frag& f, const unsigned short* base, int ld,
                                        int nrow0, int k0) {
    int lane = threadIdx.x & 31;
    int n = lane & 15, half = lane >> 4;
    const unsigned short* p = base + (size_t)(nrow0 + n) * ld + k0;
#pragma unroll
    for (int j = 0; j < 8; ++j) {
        int kk = (j < 4) ? (8 * half + 2 * j) : (16 + 8 * half + 2 * (j - 4));
        f.u[j] = *(const unsigned int*)(p + kk);
    }
}

// ---------------------------------------------------------------- prep (f32 -> bf16)
__global__ void prep_kernel(const float* __restrict__ tgt, const float* __restrict__ memory,
                            const float* __restrict__ pos, const float* __restrict__ qpos,
                            const float* __restrict__ Wq, const float* __restrict__ Wk,
                            const float* __restrict__ Wv, const float* __restrict__ Wfc,
                            unsigned short* q_bf, unsigned short* k_bf, unsigned short* v_bf,
                            unsigned short* wq_bf, unsigned short* wk_bf,
                            unsigned short* wv_bf, unsigned short* wfc_bf) {
    int tid = blockIdx.x * blockDim.x + threadIdx.x;
    int nth = gridDim.x * blockDim.x;
    for (int i = tid; i < BATCH * LEN_Q * D_MODEL; i += nth) {
        int d = i & 255, m = (i >> 8) & (LEN_Q - 1), b = i >> 18;
        int src = (m * BATCH + b) * D_MODEL + d;
        q_bf[i] = f2bf(tgt[src] + qpos[src]);
    }
    for (int i = tid; i < BATCH * LEN_K * D_MODEL; i += nth) {
        int d = i & 255, n = (i >> 8) & (LEN_K - 1), b = i >> 20;
        int src = (n * BATCH + b) * D_MODEL + d;
        float mm = memory[src];
        k_bf[i] = f2bf(mm + pos[src]);
        v_bf[i] = f2bf(mm);
    }
    for (int i = tid; i < N_HEAD * D_MODEL * D_MODEL; i += nth) {
        wq_bf[i]  = f2bf(Wq[i]);
        wk_bf[i]  = f2bf(Wk[i]);
        wv_bf[i]  = f2bf(Wv[i]);
        wfc_bf[i] = f2bf(Wfc[i]);
    }
}

// ------------------------------------------------- per-head projection GEMM (WMMA)
__global__ void proj_kernel(const unsigned short* __restrict__ X,
                            const unsigned short* __restrict__ W,
                            unsigned short* __restrict__ out,
                            int L, float scale, int transposed) {
    int wid = (blockIdx.x * blockDim.x + threadIdx.x) >> 5;
    int et = wid & 15;
    int mt = (wid >> 4) % (L >> 4);
    int h  = (wid / (16 * (L >> 4))) & (N_HEAD - 1);
    int b  = wid / (16 * (L >> 4) * N_HEAD);
    int m0 = mt << 4, n0 = et << 4;
    const unsigned short* Xb = X + (size_t)b * L * D_MODEL;
    const unsigned short* Wh = W + (size_t)h * D_MODEL * D_MODEL;
    v8f acc = {0.f, 0.f, 0.f, 0.f, 0.f, 0.f, 0.f, 0.f};
#pragma unroll
    for (int c = 0; c < 8; ++c) {
        Frag a, bm;
        load_a(a, Xb, D_MODEL, m0, 32 * c);
        load_bt(bm, Wh, D_MODEL, n0, 32 * c);
        acc = wmma_bf16(a, bm, acc);
    }
    int lane = threadIdx.x & 31;
    int n = lane & 15, half = lane >> 4;
    if (!transposed) {
        unsigned short* o = out + (size_t)(b * N_HEAD + h) * L * D_MODEL;
#pragma unroll
        for (int j = 0; j < 8; ++j)
            o[(size_t)(m0 + j + 8 * half) * D_MODEL + n0 + n] = f2bf(acc[j] * scale);
    } else {
        unsigned short* o = out + ((size_t)(b * N_HEAD + h) * D_MODEL + n0 + n) * L
                                + m0 + 8 * half;
        v8us pk;
#pragma unroll
        for (int j = 0; j < 8; ++j) pk[j] = f2bf(acc[j] * scale);
        *(v8us*)o = pk;
    }
}

// ------------------- fused flash-attention + head-mean of logits (4 waves = 4 heads)
// K tiles staged into LDS by the Tensor Data Mover (one 2-D D# per wave per chunk).
__global__ __launch_bounds__(128) void attn_kernel(
    const unsigned short* __restrict__ qh, const unsigned short* __restrict__ kh,
    const unsigned short* __restrict__ vhT, unsigned short* __restrict__ oc,
    float* __restrict__ attn_mean) {
    __shared__ __align__(16) unsigned short kbuf[N_HEAD][32][D_MODEL]; // 64 KB TDM dest
    __shared__ float smean[N_HEAD][16][32];                            // 8 KB
    __shared__ __align__(16) unsigned short plds[N_HEAD][16][32];      // 4 KB

    int b  = blockIdx.x >> 6;
    int q0 = (blockIdx.x & 63) << 4;
    int w  = threadIdx.x >> 5;                 // wave == head
    int wu = __builtin_amdgcn_readfirstlane(w);
    int lane = threadIdx.x & 31;
    int n = lane & 15, half = lane >> 4;

    const unsigned short* qhb = qh  + (size_t)(b * N_HEAD + wu) * LEN_Q * D_MODEL;
    const unsigned short* khb = kh  + (size_t)(b * N_HEAD + wu) * LEN_K * D_MODEL;
    const unsigned short* vtb = vhT + (size_t)(b * N_HEAD + wu) * D_MODEL * LEN_K;

#if HAVE_TDM
    unsigned long long kh_base = (unsigned long long)(uintptr_t)khb;
    unsigned int lds_base = (unsigned int)(uintptr_t)&kbuf[wu][0][0]; // LDS byte offset
    // D# group1: data_size=2B; tensor_dim0=256, tensor_dim1=32; tile 256x32; stride0=256
    v8i g1;
    g1[0] = 0x00010000;        // data_size=1 (2 bytes)
    g1[1] = (int)(256u << 16); // tensor_dim0[15:0]
    g1[2] = (int)(32u << 16);  // tensor_dim1[15:0]
    g1[3] = (int)(256u << 16); // tile_dim0
    g1[4] = 32;                // tile_dim1
    g1[5] = 256;               // tensor_dim0_stride[31:0]
    g1[6] = 0;
    g1[7] = 0;
    v4i gz  = {0, 0, 0, 0};                    // groups 2/3 unused (2-D tensor)
    v8i gz8 = {0, 0, 0, 0, 0, 0, 0, 0};        // trailing zero group (6-arg builtin)
#endif

    Frag qf[8];                                 // resident Q tile (16 x 256)
#pragma unroll
    for (int c = 0; c < 8; ++c) load_a(qf[c], qhb, D_MODEL, q0, 32 * c);

    v8f accv[16];
    v8f zero = {0.f, 0.f, 0.f, 0.f, 0.f, 0.f, 0.f, 0.f};
#pragma unroll
    for (int t = 0; t < 16; ++t) accv[t] = zero;
    float m_run[8], l_run[8];
#pragma unroll
    for (int j = 0; j < 8; ++j) { m_run[j] = -1e30f; l_run[j] = 0.f; }

    for (int k0 = 0; k0 < LEN_K; k0 += 32) {
#if HAVE_TDM
        // DMA this chunk's K tile (32 keys x 256) into LDS via the Tensor Data Mover
        unsigned long long ga = kh_base + (unsigned long long)k0 * (D_MODEL * 2);
        v4u g0;
        g0[0] = 1u;                      // count=1, user descriptor
        g0[1] = lds_base;                // lds_addr
        g0[2] = (unsigned int)ga;        // global_addr[31:0]
        g0[3] = (unsigned int)(ga >> 32) | 0x80000000u; // global_addr[56:32] | type=2
        asm volatile("s_wait_dscnt 0x0" ::: "memory");  // WAR vs last chunk's LDS reads
        __builtin_amdgcn_tensor_load_to_lds(g0, g1, gz, gz, gz8, 0);
#if __has_builtin(__builtin_amdgcn_s_wait_tensorcnt)
        __builtin_amdgcn_s_wait_tensorcnt(0);
#else
        asm volatile("s_wait_tensorcnt 0x0" ::: "memory");
#endif
        const unsigned short* ksrc = &kbuf[wu][0][0];
        const int krow0 = 0, kld = D_MODEL;
#else
        const unsigned short* ksrc = khb;
        const int krow0 = k0, kld = D_MODEL;
#endif
        __builtin_prefetch(vtb + (size_t)(lane * 8) * LEN_K + k0 + 32, 0, 1);
        // S = (Q*scale) · K^T   (scale pre-folded into qh)
        v8f s0 = zero, s1 = zero;
#pragma unroll
        for (int c = 0; c < 8; ++c) {
            Frag bk;
            load_bt(bk, ksrc, kld, krow0, 32 * c);
            s0 = wmma_bf16(qf[c], bk, s0);
            load_bt(bk, ksrc, kld, krow0 + 16, 32 * c);
            s1 = wmma_bf16(qf[c], bk, s1);
        }
        // stage this head's logits (pre-scaled by 1/H) for the cross-head mean
#pragma unroll
        for (int j = 0; j < 8; ++j) {
            smean[w][j + 8 * half][n]      = 0.25f * s0[j];
            smean[w][j + 8 * half][16 + n] = 0.25f * s1[j];
        }
        // online softmax
        float alpha[8];
#pragma unroll
        for (int j = 0; j < 8; ++j) {
            float r = fmaxf(s0[j], s1[j]);
#pragma unroll
            for (int d = 1; d < 16; d <<= 1) r = fmaxf(r, __shfl_xor(r, d, 32));
            float mn = fmaxf(m_run[j], r);
            alpha[j] = __expf(m_run[j] - mn);
            float p0 = __expf(s0[j] - mn);
            float p1 = __expf(s1[j] - mn);
            float rs = p0 + p1;
#pragma unroll
            for (int d = 1; d < 16; d <<= 1) rs += __shfl_xor(rs, d, 32);
            l_run[j] = l_run[j] * alpha[j] + rs;
            m_run[j] = mn;
            plds[w][j + 8 * half][n]      = f2bf(p0);
            plds[w][j + 8 * half][16 + n] = f2bf(p1);
        }
        __syncthreads();
        // cooperative mean over 4 heads -> attn_mean output (coalesced, no atomics)
#pragma unroll
        for (int rep = 0; rep < 4; ++rep) {
            int idx = threadIdx.x + rep * 128;           // 0..511
            int r = idx >> 5, c2 = idx & 31;
            float s = smean[0][r][c2] + smean[1][r][c2] + smean[2][r][c2] + smean[3][r][c2];
            attn_mean[(size_t)(b * LEN_Q + q0 + r) * LEN_K + k0 + c2] = s;
        }
        Frag pf;                                          // P tile C-layout -> A-layout via LDS
        load_a(pf, &plds[w][0][0], 32, 0, 0);
        __syncthreads();
        // O = O*alpha + P · V   (V pre-transposed -> contiguous B-fragment loads)
#pragma unroll
        for (int t = 0; t < 16; ++t) {
#pragma unroll
            for (int j = 0; j < 8; ++j) accv[t][j] *= alpha[j];
            Frag vb;
            load_bt(vb, vtb, LEN_K, 16 * t, k0);
            accv[t] = wmma_bf16(pf, vb, accv[t]);
        }
    }
    // normalize + store concatenated heads [B][Lq][H*D] as bf16
    float inv[8];
#pragma unroll
    for (int j = 0; j < 8; ++j) inv[j] = 1.0f / l_run[j];
#pragma unroll
    for (int t = 0; t < 16; ++t) {
#pragma unroll
        for (int j = 0; j < 8; ++j) {
            int row = q0 + j + 8 * half;
            oc[(size_t)(b * LEN_Q + row) * (N_HEAD * D_MODEL) + w * D_MODEL + 16 * t + n] =
                f2bf(accv[t][j] * inv[j]);
        }
    }
}

// ---------------------------------------------------------- fc GEMM (K=1024, WMMA)
__global__ void fc_kernel(const unsigned short* __restrict__ oc,
                          const unsigned short* __restrict__ wfc,
                          float* __restrict__ tgt2) {
    int wid = (blockIdx.x * blockDim.x + threadIdx.x) >> 5;
    int nt = wid & 15;
    int mt = (wid >> 4) & 63;
    int b  = wid >> 10;
    int m0 = mt << 4, n0 = nt << 4;
    const unsigned short* A = oc + (size_t)b * LEN_Q * (N_HEAD * D_MODEL);
    v8f acc = {0.f, 0.f, 0.f, 0.f, 0.f, 0.f, 0.f, 0.f};
#pragma unroll
    for (int c = 0; c < 32; ++c) {
        Frag a, bm;
        load_a(a, A, N_HEAD * D_MODEL, m0, 32 * c);
        load_bt(bm, wfc, N_HEAD * D_MODEL, n0, 32 * c);
        acc = wmma_bf16(a, bm, acc);
    }
    int lane = threadIdx.x & 31;
    int n = lane & 15, half = lane >> 4;
#pragma unroll
    for (int j = 0; j < 8; ++j)
        tgt2[((size_t)b * LEN_Q + m0 + j + 8 * half) * D_MODEL + n0 + n] = acc[j];
}

// ---------------------------------------------- residual + LayerNorm (row of 256)
__global__ void ln_kernel(const float* __restrict__ tgt, const float* __restrict__ tgt2,
                          const float* __restrict__ g, const float* __restrict__ beta,
                          float* __restrict__ y) {
    __shared__ float red[2][8];
    int mb = blockIdx.x;               // m*8 + b
    int m = mb >> 3, b = mb & 7;
    int d = threadIdx.x;
    float x = tgt[(size_t)mb * D_MODEL + d] + tgt2[((size_t)b * LEN_Q + m) * D_MODEL + d];
    float s = x, sq = x * x;
#pragma unroll
    for (int o = 1; o < 32; o <<= 1) { s += __shfl_xor(s, o, 32); sq += __shfl_xor(sq, o, 32); }
    int wv = threadIdx.x >> 5, ln = threadIdx.x & 31;
    if (ln == 0) { red[0][wv] = s; red[1][wv] = sq; }
    __syncthreads();
    if (threadIdx.x < 32) {
        float a = (ln < 8) ? red[0][ln] : 0.f;
        float c = (ln < 8) ? red[1][ln] : 0.f;
#pragma unroll
        for (int o = 1; o < 8; o <<= 1) { a += __shfl_xor(a, o, 32); c += __shfl_xor(c, o, 32); }
        if (ln == 0) { red[0][0] = a; red[1][0] = c; }
    }
    __syncthreads();
    float mu  = red[0][0] * (1.0f / D_MODEL);
    float var = red[1][0] * (1.0f / D_MODEL) - mu * mu;
    float r = rsqrtf(var + 1e-5f);
    y[(size_t)mb * D_MODEL + d] = (x - mu) * r * g[d] + beta[d];
}

extern "C" void kernel_launch(void* const* d_in, const int* in_sizes, int n_in,
                              void* d_out, int out_size, void* d_ws, size_t ws_size,
                              hipStream_t stream) {
    (void)in_sizes; (void)n_in; (void)out_size; (void)ws_size;
    const float* tgt    = (const float*)d_in[0];
    const float* memory = (const float*)d_in[1];
    const float* pos    = (const float*)d_in[2];
    const float* qpos   = (const float*)d_in[3];
    const float* Wq     = (const float*)d_in[4];
    const float* Wk     = (const float*)d_in[5];
    const float* Wv     = (const float*)d_in[6];
    const float* Wfc    = (const float*)d_in[7];
    const float* ln_g   = (const float*)d_in[8];
    const float* ln_b   = (const float*)d_in[9];

    char* ws = (char*)d_ws;
    size_t off = 0;
    auto alloc = [&](size_t bytes) {
        void* p = ws + off;
        off = (off + bytes + 255) & ~(size_t)255;
        return p;
    };
    unsigned short* q_bf   = (unsigned short*)alloc((size_t)BATCH * LEN_Q * D_MODEL * 2);
    unsigned short* k_bf   = (unsigned short*)alloc((size_t)BATCH * LEN_K * D_MODEL * 2);
    unsigned short* v_bf   = (unsigned short*)alloc((size_t)BATCH * LEN_K * D_MODEL * 2);
    unsigned short* wq_bf  = (unsigned short*)alloc((size_t)N_HEAD * D_MODEL * D_MODEL * 2);
    unsigned short* wk_bf  = (unsigned short*)alloc((size_t)N_HEAD * D_MODEL * D_MODEL * 2);
    unsigned short* wv_bf  = (unsigned short*)alloc((size_t)N_HEAD * D_MODEL * D_MODEL * 2);
    unsigned short* wfc_bf = (unsigned short*)alloc((size_t)D_MODEL * N_HEAD * D_MODEL * 2);
    unsigned short* qh_bf  = (unsigned short*)alloc((size_t)BATCH * N_HEAD * LEN_Q * D_MODEL * 2);
    unsigned short* kh_bf  = (unsigned short*)alloc((size_t)BATCH * N_HEAD * LEN_K * D_MODEL * 2);
    unsigned short* vhT_bf = (unsigned short*)alloc((size_t)BATCH * N_HEAD * D_MODEL * LEN_K * 2);
    unsigned short* oc_bf  = (unsigned short*)alloc((size_t)BATCH * LEN_Q * N_HEAD * D_MODEL * 2);
    float*          tgt2   = (float*)alloc((size_t)BATCH * LEN_Q * D_MODEL * 4);

    float* y_out    = (float*)d_out;                                   // [Lq,B,D]
    float* attn_out = (float*)d_out + (size_t)LEN_Q * BATCH * D_MODEL; // [B,1,Lq,Lk]

    prep_kernel<<<2048, 256, 0, stream>>>(tgt, memory, pos, qpos, Wq, Wk, Wv, Wfc,
                                          q_bf, k_bf, v_bf, wq_bf, wk_bf, wv_bf, wfc_bf);

    const float scale = 1.0f / 16.0f;  // 1/sqrt(256), folded into qh
    proj_kernel<<<4096, 256, 0, stream>>>(q_bf, wq_bf, qh_bf, LEN_Q, scale, 0);
    proj_kernel<<<16384, 256, 0, stream>>>(k_bf, wk_bf, kh_bf, LEN_K, 1.0f, 0);
    proj_kernel<<<16384, 256, 0, stream>>>(v_bf, wv_bf, vhT_bf, LEN_K, 1.0f, 1);

    attn_kernel<<<BATCH * (LEN_Q / 16), 128, 0, stream>>>(qh_bf, kh_bf, vhT_bf, oc_bf, attn_out);

    fc_kernel<<<1024, 256, 0, stream>>>(oc_bf, wfc_bf, tgt2);

    ln_kernel<<<LEN_Q * BATCH, 256, 0, stream>>>(tgt, tgt2, ln_g, ln_b, y_out);
}
